// RWKV6TimeMix_69441031242174
// MI455X (gfx1250) — matrix operand order repaired and checked by reading, not compiled, not run on hardware
//
#include <hip/hip_runtime.h>
#include <hip/hip_bf16.h>

// RWKV6 TimeMix for MI455X (gfx1250, wave32).
//   - bf16 WMMA GEMMs (f32 accumulate) for all matmuls (compute-bound: ~344 GFLOP
//     across the five DxD projections at AI ~480 flop/byte)
//   - Tensor Data Mover (TDM) streams GEMM tiles into LDS, double-buffered,
//     overlapped with WMMA; TENSORcnt + barrier handoff. Falls back to
//     global->VGPR->LDS staging when the TDM builtin is unavailable.
//   - epilogues fused: tanh / sigmoid / silu / exp(-exp(+decay)) / token-shift mixing
//   - WKV scan: one 64-thread block per (b,h), state resident in VGPRs
//   - groupnorm+gate fused, final W_o projection via WMMA

typedef __attribute__((ext_vector_type(16))) __bf16 v16bf;
typedef __attribute__((ext_vector_type(8)))  __bf16 v8bf;
typedef __attribute__((ext_vector_type(8)))  float  v8f;
typedef __attribute__((ext_vector_type(4)))  unsigned int v4u_t;
typedef __attribute__((ext_vector_type(8)))  int v8i_t;
typedef __attribute__((ext_vector_type(4)))  int v4i_t;

enum { EPI_NONE = 0, EPI_TANH, EPI_SIGMOID, EPI_SILU, EPI_DECAY, EPI_MIX };

constexpr int LDSS = 40;  // bf16 elems per LDS row: 64B data + 16B pad (TDM pad: 16DW + 4DW)

// ---------------- TDM support (probe-verified builtins, toolchain-hedged) ----
#if defined(__HIP_DEVICE_COMPILE__) && __has_builtin(__builtin_amdgcn_tensor_load_to_lds) && \
    __has_builtin(__builtin_amdgcn_s_wait_tensorcnt)
#define HAVE_TDM 1
#else
#define HAVE_TDM 0
#endif

#if HAVE_TDM
__device__ __forceinline__ void tdm_issue(v4u_t g0, v8i_t g1) {
#if __clang_major__ >= 23
  __builtin_amdgcn_tensor_load_to_lds(g0, g1, (v4i_t)0, (v4i_t)0, (v8i_t)0, 0);
#else
  __builtin_amdgcn_tensor_load_to_lds(g0, g1, (v4i_t)0, (v4i_t)0, 0);
#endif
}

__device__ __forceinline__ unsigned lds_addr_of(const void* p) {
  return (unsigned)(unsigned long long)(__attribute__((address_space(3))) const void*)p;
}

// 2-D tile load: tile_dim0=32 bf16 (contiguous, 64B rows), tile_dim1=128 rows,
// hardware pads +16B per row into LDS to match the LDSS=40 pitch.
__device__ __forceinline__ void tdm_load_tile(unsigned lds_off, const __bf16* gptr,
                                              unsigned tensor_d0, unsigned tensor_d1,
                                              unsigned stride0) {
  const unsigned long long ga = (unsigned long long)gptr;
  v4u_t g0;
  g0[0] = 1u;                                             // count=1 (valid D#)
  g0[1] = lds_off;                                        // lds_addr (bytes)
  g0[2] = (unsigned)ga;                                   // global_addr[31:0]
  g0[3] = ((unsigned)(ga >> 32) & 0x01ffffffu) | (2u << 30);  // addr[56:32] | type=2
  v8i_t g1;
  g1[0] = (int)((1u << 16)      // data_size = 2 bytes
              | (1u << 20)      // pad_enable
              | (3u << 22)      // pad_interval: 16 DWORDs (one 64B tile row)
              | (3u << 25));    // pad_amount: 4 DWORDs (16B)
  g1[1] = (int)((tensor_d0 & 0xffffu) << 16);                       // tensor_dim0 lo
  g1[2] = (int)((tensor_d0 >> 16) | ((tensor_d1 & 0xffffu) << 16)); // d0 hi | d1 lo
  g1[3] = (int)((tensor_d1 >> 16) | (32u << 16));                   // d1 hi | tile_dim0=32
  g1[4] = (int)128;                                                 // tile_dim1=128
  g1[5] = (int)stride0;                                             // dim0 stride (elems)
  g1[6] = 0;
  g1[7] = 0;
  tdm_issue(g0, g1);
}
#endif

// ---- LDS -> WMMA fragment loads (wave32 layouts per CDNA5 ISA 7.12.2) ------
__device__ __forceinline__ v16bf load_frag_a(const __bf16* As, int rowBase, int lane) {
  const int r  = rowBase + (lane & 15);
  const int ks = (lane >> 4) * 8;
  const v8bf lo = *(const v8bf*)(As + r * LDSS + ks);
  const v8bf hi = *(const v8bf*)(As + r * LDSS + ks + 16);
  return __builtin_shufflevector(lo, hi, 0,1,2,3,4,5,6,7,8,9,10,11,12,13,14,15);
}
__device__ __forceinline__ v16bf load_frag_b(const __bf16* Bs, int rowBase, int lane) {
  const int r  = rowBase + (lane & 15);
  const int ks = (lane >> 4) * 8;
  const v8bf lo = *(const v8bf*)(Bs + r * LDSS + ks);
  const v8bf hi = *(const v8bf*)(Bs + r * LDSS + ks + 16);
  return __builtin_shufflevector(lo, hi, 0,1,2,3,4,5,6,7,8,9,10,11,12,13,14,15);
}

// ---- Generic WMMA GEMM: C(MxN) = epi(A(MxK bf16) * Bt(NxK bf16)^T) ----------
// 256 threads = 8 waves (2x4); block tile 128x128; k-tile 32; wave tile 64x32.
template <int EPI, int OUTBF>
__global__ __launch_bounds__(256) void wmma_gemm(
    const __bf16* __restrict__ A, int lda,
    const __bf16* __restrict__ Bt, int ldb,
    void* __restrict__ Cv, int ldc,
    int M, int N, int K,
    const float* __restrict__ bias,   // EPI_DECAY: per-col time_decay
    const float* __restrict__ xin,    // EPI_MIX:   x (BT x D)
    const float* __restrict__ prevp,  // EPI_MIX:   prev (B x D)
    const float* __restrict__ maa,    // EPI_MIX:   per-col maa
    int T, int D) {
  __shared__ __align__(16) __bf16 As[2][128 * LDSS];
  __shared__ __align__(16) __bf16 Bs[2][128 * LDSS];

  const int tid  = threadIdx.x;
  const int lane = tid & 31;
  const int wave = tid >> 5;
  const int wm   = wave >> 2;  // 0..1
  const int wn   = wave & 3;   // 0..3
  const int blockM = blockIdx.y * 128;
  const int blockN = blockIdx.x * 128;

  v8f acc[4][2] = {};

#if HAVE_TDM
  // ---- TDM double-buffered pipeline: wave0 issues descriptors, DMA fills LDS
  const __bf16* aTile = A  + (size_t)blockM * lda;
  const __bf16* bTile = Bt + (size_t)blockN * ldb;
  if (wave == 0) {
    tdm_load_tile(lds_addr_of(As[0]), aTile, (unsigned)K, (unsigned)M, (unsigned)lda);
    tdm_load_tile(lds_addr_of(Bs[0]), bTile, (unsigned)K, (unsigned)N, (unsigned)ldb);
  }
  int buf = 0;
  for (int k0 = 0; k0 < K; k0 += 32) {
    if (wave == 0) __builtin_amdgcn_s_wait_tensorcnt(0);  // current tiles landed
    __syncthreads();
    if (wave == 0 && k0 + 32 < K) {                        // prefetch next tiles
      tdm_load_tile(lds_addr_of(As[buf ^ 1]), aTile + k0 + 32,
                    (unsigned)K, (unsigned)M, (unsigned)lda);
      tdm_load_tile(lds_addr_of(Bs[buf ^ 1]), bTile + k0 + 32,
                    (unsigned)K, (unsigned)N, (unsigned)ldb);
    }
    const __bf16* Ac = As[buf];
    const __bf16* Bc = Bs[buf];
    v16bf af[4], bfr[2];
#pragma unroll
    for (int mi = 0; mi < 4; ++mi) af[mi]  = load_frag_a(Ac, wm * 64 + mi * 16, lane);
#pragma unroll
    for (int ni = 0; ni < 2; ++ni) bfr[ni] = load_frag_b(Bc, wn * 32 + ni * 16, lane);
#pragma unroll
    for (int mi = 0; mi < 4; ++mi)
#pragma unroll
      for (int ni = 0; ni < 2; ++ni)
        acc[mi][ni] = __builtin_amdgcn_wmma_f32_16x16x32_bf16(
            false, af[mi], false, bfr[ni], (short)0, acc[mi][ni], false, false);
    __syncthreads();  // all waves done with buf before TDM may overwrite it
    buf ^= 1;
  }
#else
  // ---- fallback: cooperative global->VGPR->LDS staging
  const int lrow = tid >> 1;
  const int lcol = (tid & 1) * 16;
  const int aRow = (blockM + lrow < M) ? (blockM + lrow) : (M - 1);
  const int bRow = (blockN + lrow < N) ? (blockN + lrow) : (N - 1);
  const __bf16* aSrc = A  + (size_t)aRow * lda + lcol;
  const __bf16* bSrc = Bt + (size_t)bRow * ldb + lcol;
  __bf16* aDst = As[0] + lrow * LDSS + lcol;
  __bf16* bDst = Bs[0] + lrow * LDSS + lcol;
  for (int k0 = 0; k0 < K; k0 += 32) {
    const v8bf a0 = *(const v8bf*)(aSrc + k0);
    const v8bf a1 = *(const v8bf*)(aSrc + k0 + 8);
    const v8bf b0 = *(const v8bf*)(bSrc + k0);
    const v8bf b1 = *(const v8bf*)(bSrc + k0 + 8);
    if (k0 + 32 < K) {
      __builtin_prefetch(aSrc + k0 + 32, 0, 3);
      __builtin_prefetch(bSrc + k0 + 32, 0, 3);
    }
    __syncthreads();
    *(v8bf*)aDst = a0;  *(v8bf*)(aDst + 8) = a1;
    *(v8bf*)bDst = b0;  *(v8bf*)(bDst + 8) = b1;
    __syncthreads();
    v16bf af[4], bfr[2];
#pragma unroll
    for (int mi = 0; mi < 4; ++mi) af[mi]  = load_frag_a(As[0], wm * 64 + mi * 16, lane);
#pragma unroll
    for (int ni = 0; ni < 2; ++ni) bfr[ni] = load_frag_b(Bs[0], wn * 32 + ni * 16, lane);
#pragma unroll
    for (int mi = 0; mi < 4; ++mi)
#pragma unroll
      for (int ni = 0; ni < 2; ++ni)
        acc[mi][ni] = __builtin_amdgcn_wmma_f32_16x16x32_bf16(
            false, af[mi], false, bfr[ni], (short)0, acc[mi][ni], false, false);
  }
#endif

  // epilogue: C/D layout — VGPR j holds M = 8*(lane>>4)+j, N = lane&15 per 16x16 tile
  float*  Cf = (float*)Cv;
  __bf16* Cb = (__bf16*)Cv;
#pragma unroll
  for (int mi = 0; mi < 4; ++mi) {
#pragma unroll
    for (int ni = 0; ni < 2; ++ni) {
#pragma unroll
      for (int j = 0; j < 8; ++j) {
        const int row = blockM + wm * 64 + mi * 16 + ((lane >> 4) << 3) + j;
        const int col = blockN + wn * 32 + ni * 16 + (lane & 15);
        if (row < M && col < N) {
          float v = acc[mi][ni][j];
          if (EPI == EPI_TANH)         v = tanhf(v);
          else if (EPI == EPI_SIGMOID) v = 1.f / (1.f + __expf(-v));
          else if (EPI == EPI_SILU)    v = v / (1.f + __expf(-v));
          else if (EPI == EPI_DECAY)   { v += bias[col]; v = __expf(-__expf(v)); }
          else if (EPI == EPI_MIX) {
            const int t = row % T;
            const int b = row / T;
            const size_t xi = (size_t)row * D + col;
            const float xv = xin[xi];
            const float sh = t ? xin[xi - D] : prevp[(size_t)b * D + col];
            v = xv + (sh - xv) * (maa[col] + v);
          }
          if (OUTBF) Cb[(size_t)row * ldc + col] = (__bf16)v;
          else       Cf[(size_t)row * ldc + col] = v;
        }
      }
    }
  }
}

// ---- weight pack: W (KxN fp32 row-major) -> Wt (NxK bf16 row-major) ---------
__global__ __launch_bounds__(256) void pack_wt(const float* __restrict__ W,
                                               __bf16* __restrict__ Wt, int K, int N) {
  __shared__ float tile[32][33];
  const int n0 = blockIdx.x * 32, k0 = blockIdx.y * 32;
  const int tx = threadIdx.x & 31;
  const int ty = threadIdx.x >> 5;
#pragma unroll
  for (int i = 0; i < 4; ++i) {
    const int k = k0 + ty + i * 8, n = n0 + tx;
    tile[ty + i * 8][tx] = (k < K && n < N) ? W[(size_t)k * N + n] : 0.f;
  }
  __syncthreads();
#pragma unroll
  for (int i = 0; i < 4; ++i) {
    const int n = n0 + ty + i * 8, k = k0 + tx;
    if (n < N && k < K) Wt[(size_t)n * K + k] = (__bf16)tile[tx][ty + i * 8];
  }
}

// ---- xxx = x + (shift(x)-x)*x_maa, bf16 out ---------------------------------
__global__ __launch_bounds__(256) void shift_mix_x(const float* __restrict__ x,
                                                   const float* __restrict__ prev,
                                                   const float* __restrict__ x_maa,
                                                   __bf16* __restrict__ xxx, int T, int D) {
  const size_t idx = (size_t)blockIdx.x * 256 + threadIdx.x;
  const int d  = (int)(idx % D);
  const size_t bt = idx / D;
  const int t  = (int)(bt % T);
  const int b  = (int)(bt / T);
  const float xv = x[idx];
  const float sh = t ? x[idx - D] : prev[(size_t)b * D + d];
  xxx[idx] = (__bf16)(xv + (sh - xv) * x_maa[d]);
}

// ---- WKV sequential scan: one 64-thread block per (b,h); lane owns column v -
__global__ __launch_bounds__(64) void wkv_scan(
    const float* __restrict__ r, const float* __restrict__ k,
    const float* __restrict__ v, const float* __restrict__ w,
    const float* __restrict__ u, const float* __restrict__ st_in,
    float* __restrict__ outp, float* __restrict__ st_out, int B, int T, int H) {
  const int bh = blockIdx.x;
  const int b = bh / H, h = bh % H;
  const int vv = threadIdx.x;  // 0..63
  __shared__ float rs[64], ks[64], ws[64], us[64];
  float st[64];
  us[vv] = u[h * 64 + vv];
#pragma unroll
  for (int kk = 0; kk < 64; ++kk)
    st[kk] = st_in[((size_t)(b * H + h) * 64 + kk) * 64 + vv];
  __syncthreads();
  const size_t base = ((size_t)b * T * H + h) * 64;
  const size_t step = (size_t)H * 64;
  for (int t = 0; t < T; ++t) {
    const size_t o = base + (size_t)t * step;
    rs[vv] = r[o + vv];
    ks[vv] = k[o + vv];
    ws[vv] = w[o + vv];
    __syncthreads();
    const float vt = v[o + vv];
    float acc = 0.f;
#pragma unroll
    for (int kk = 0; kk < 64; ++kk) {
      const float kvv = ks[kk] * vt;
      acc = fmaf(rs[kk], fmaf(us[kk], kvv, st[kk]), acc);
      st[kk] = fmaf(st[kk], ws[kk], kvv);
    }
    outp[o + vv] = acc;
    __syncthreads();
  }
#pragma unroll
  for (int kk = 0; kk < 64; ++kk)
    st_out[((size_t)(b * H + h) * 64 + kk) * 64 + vv] = st[kk];
}

// ---- per-(bt,h) groupnorm over HS=64, * ln_g + ln_b, * g -> bf16 ------------
__global__ __launch_bounds__(64) void gnorm_gate(const float* __restrict__ wkv,
                                                 const float* __restrict__ g,
                                                 const float* __restrict__ ln_g,
                                                 const float* __restrict__ ln_b,
                                                 __bf16* __restrict__ a_o, int H) {
  const int bt = blockIdx.x, h = blockIdx.y, vv = threadIdx.x;
  __shared__ float sv[64];
  const size_t o = ((size_t)bt * H + h) * 64;
  const float xx = wkv[o + vv];
  sv[vv] = xx;
  __syncthreads();
  float mu = 0.f;
#pragma unroll 8
  for (int i = 0; i < 64; ++i) mu += sv[i];
  mu *= (1.f / 64.f);
  float m2 = 0.f;
#pragma unroll 8
  for (int i = 0; i < 64; ++i) { const float d = sv[i] - mu; m2 = fmaf(d, d, m2); }
  m2 *= (1.f / 64.f);
  const float xn = (xx - mu) * rsqrtf(m2 + 1e-5f);
  const int d = h * 64 + vv;
  const size_t di = (size_t)bt * (H * 64) + d;
  a_o[di] = (__bf16)((xn * ln_g[d] + ln_b[d]) * g[di]);
}

__global__ __launch_bounds__(256) void xlast_copy(const float* __restrict__ x,
                                                  float* __restrict__ o, int T, int D) {
  const int idx = blockIdx.x * 256 + threadIdx.x;
  const int b = idx / D, d = idx % D;
  o[idx] = x[((size_t)b * T + (T - 1)) * D + d];
}

extern "C" void kernel_launch(void* const* d_in, const int* in_sizes, int n_in,
                              void* d_out, int out_size, void* d_ws, size_t ws_size,
                              hipStream_t stream) {
  (void)in_sizes; (void)n_in; (void)out_size; (void)ws_size;
  const int B = 4, T = 2048, D = 2048, H = 32, E5 = 160, TD = 64;
  const int BT = B * T;

  const float* x        = (const float*)d_in[0];
  const float* prev     = (const float*)d_in[1];
  const float* state_in = (const float*)d_in[2];
  const float* x_maa    = (const float*)d_in[3];
  const float* w_maa    = (const float*)d_in[4];
  const float* k_maa    = (const float*)d_in[5];
  const float* v_maa    = (const float*)d_in[6];
  const float* r_maa    = (const float*)d_in[7];
  const float* g_maa    = (const float*)d_in[8];
  const float* tm_w1    = (const float*)d_in[9];
  const float* tm_w2    = (const float*)d_in[10];
  const float* td_w1    = (const float*)d_in[11];
  const float* td_w2    = (const float*)d_in[12];
  const float* time_decay = (const float*)d_in[13];
  const float* time_first = (const float*)d_in[14];
  const float* W_r = (const float*)d_in[15];
  const float* W_k = (const float*)d_in[16];
  const float* W_v = (const float*)d_in[17];
  const float* W_g = (const float*)d_in[18];
  const float* W_o = (const float*)d_in[19];
  const float* ln_g = (const float*)d_in[20];
  const float* ln_b = (const float*)d_in[21];

  float* out       = (float*)d_out;                 // (B,T,D)
  float* x_last    = out + (size_t)BT * D;          // (B,D)
  float* state_out = x_last + (size_t)B * D;        // (B,H,HS,HS)

  char* ws = (char*)d_ws;
  size_t off = 0;
  auto alloc = [&](size_t bytes) -> void* {
    void* p = ws + off;
    off = (off + bytes + 255) & ~(size_t)255;
    return p;
  };
  __bf16* wt_r   = (__bf16*)alloc((size_t)D * D * 2);
  __bf16* wt_k   = (__bf16*)alloc((size_t)D * D * 2);
  __bf16* wt_v   = (__bf16*)alloc((size_t)D * D * 2);
  __bf16* wt_g   = (__bf16*)alloc((size_t)D * D * 2);
  __bf16* wt_o   = (__bf16*)alloc((size_t)D * D * 2);
  __bf16* wt_tm1 = (__bf16*)alloc((size_t)E5 * D * 2);
  __bf16* wt_tm2 = (__bf16*)alloc((size_t)5 * D * 32 * 2);
  __bf16* wt_td1 = (__bf16*)alloc((size_t)TD * D * 2);
  __bf16* wt_td2 = (__bf16*)alloc((size_t)D * TD * 2);
  __bf16* xxx_bf = (__bf16*)alloc((size_t)BT * D * 2);
  __bf16* m_bf   = (__bf16*)alloc((size_t)BT * E5 * 2);
  __bf16* act[5];
  for (int f = 0; f < 5; ++f) act[f] = (__bf16*)alloc((size_t)BT * D * 2);  // wx,kx,vx,rx,gx
  __bf16* wtmp_bf = (__bf16*)alloc((size_t)BT * TD * 2);
  float* rbuf   = (float*)alloc((size_t)BT * D * 4);
  float* kbuf   = (float*)alloc((size_t)BT * D * 4);
  float* vbuf   = (float*)alloc((size_t)BT * D * 4);
  float* gbuf   = (float*)alloc((size_t)BT * D * 4);
  float* wbuf   = (float*)alloc((size_t)BT * D * 4);
  float* wkvbuf = (float*)alloc((size_t)BT * D * 4);
  __bf16* a_o   = (__bf16*)alloc((size_t)BT * D * 2);

  // 0) weight packing (fp32 -> bf16, transposed)
  auto pg = [](int N_, int K_) { return dim3((N_ + 31) / 32, (K_ + 31) / 32); };
  pack_wt<<<pg(D, D), 256, 0, stream>>>(W_r, wt_r, D, D);
  pack_wt<<<pg(D, D), 256, 0, stream>>>(W_k, wt_k, D, D);
  pack_wt<<<pg(D, D), 256, 0, stream>>>(W_v, wt_v, D, D);
  pack_wt<<<pg(D, D), 256, 0, stream>>>(W_g, wt_g, D, D);
  pack_wt<<<pg(D, D), 256, 0, stream>>>(W_o, wt_o, D, D);
  pack_wt<<<pg(E5, D), 256, 0, stream>>>(tm_w1, wt_tm1, D, E5);
  for (int f = 0; f < 5; ++f)
    pack_wt<<<pg(D, 32), 256, 0, stream>>>(tm_w2 + (size_t)f * 32 * D,
                                           wt_tm2 + (size_t)f * D * 32, 32, D);
  pack_wt<<<pg(TD, D), 256, 0, stream>>>(td_w1, wt_td1, D, TD);
  pack_wt<<<pg(D, TD), 256, 0, stream>>>(td_w2, wt_td2, TD, D);

  // 1) token-shift mix for tm path
  shift_mix_x<<<(size_t)BT * D / 256, 256, 0, stream>>>(x, prev, x_maa, xxx_bf, T, D);

  auto gemmGrid = [](int M_, int N_) { return dim3((N_ + 127) / 128, (M_ + 127) / 128); };

  // 2) m = tanh(xxx @ tm_w1) -> bf16 (BT x 160)
  wmma_gemm<EPI_TANH, 1><<<gemmGrid(BT, E5), 256, 0, stream>>>(
      xxx_bf, D, wt_tm1, D, m_bf, E5, BT, E5, D, nullptr, nullptr, nullptr, nullptr, T, D);

  // 3) act_f = x + sx*(maa_f + m_f @ tm_w2[f]) -> bf16 (mixing fused in epilogue)
  const float* maas[5] = {w_maa, k_maa, v_maa, r_maa, g_maa};
  for (int f = 0; f < 5; ++f)
    wmma_gemm<EPI_MIX, 1><<<gemmGrid(BT, D), 256, 0, stream>>>(
        m_bf + f * 32, E5, wt_tm2 + (size_t)f * D * 32, 32, act[f], D,
        BT, D, 32, nullptr, x, prev, maas[f], T, D);

  // 4) big projections
  wmma_gemm<EPI_SIGMOID, 0><<<gemmGrid(BT, D), 256, 0, stream>>>(
      act[3], D, wt_r, D, rbuf, D, BT, D, D, nullptr, nullptr, nullptr, nullptr, T, D);
  wmma_gemm<EPI_NONE, 0><<<gemmGrid(BT, D), 256, 0, stream>>>(
      act[1], D, wt_k, D, kbuf, D, BT, D, D, nullptr, nullptr, nullptr, nullptr, T, D);
  wmma_gemm<EPI_NONE, 0><<<gemmGrid(BT, D), 256, 0, stream>>>(
      act[2], D, wt_v, D, vbuf, D, BT, D, D, nullptr, nullptr, nullptr, nullptr, T, D);
  wmma_gemm<EPI_SILU, 0><<<gemmGrid(BT, D), 256, 0, stream>>>(
      act[4], D, wt_g, D, gbuf, D, BT, D, D, nullptr, nullptr, nullptr, nullptr, T, D);

  // 5) decay path: wtmp = tanh(wx @ td_w1); w = exp(-exp(wtmp @ td_w2 + decay))
  wmma_gemm<EPI_TANH, 1><<<gemmGrid(BT, TD), 256, 0, stream>>>(
      act[0], D, wt_td1, D, wtmp_bf, TD, BT, TD, D, nullptr, nullptr, nullptr, nullptr, T, D);
  wmma_gemm<EPI_DECAY, 0><<<gemmGrid(BT, D), 256, 0, stream>>>(
      wtmp_bf, TD, wt_td2, TD, wbuf, D, BT, D, TD, time_decay, nullptr, nullptr, nullptr, T, D);

  // 6) WKV scan (writes new_state straight into d_out)
  wkv_scan<<<B * H, 64, 0, stream>>>(rbuf, kbuf, vbuf, wbuf, time_first, state_in,
                                     wkvbuf, state_out, B, T, H);

  // 7) groupnorm + gate -> bf16
  gnorm_gate<<<dim3(BT, H), 64, 0, stream>>>(wkvbuf, gbuf, ln_g, ln_b, a_o, H);

  // 8) out = a_o @ W_o  (fp32 to d_out)
  wmma_gemm<EPI_NONE, 0><<<gemmGrid(BT, D), 256, 0, stream>>>(
      a_o, D, wt_o, D, out, D, BT, D, D, nullptr, nullptr, nullptr, nullptr, T, D);

  // x_last
  xlast_copy<<<(B * D) / 256, 256, 0, stream>>>(x, x_last, T, D);
}
